// MultiScaleMoE_23175643529566
// MI455X (gfx1250) — matrix operand order repaired and checked by reading, not compiled, not run on hardware
//
#include <hip/hip_runtime.h>

typedef __attribute__((ext_vector_type(16))) __bf16 v16bf;
typedef __attribute__((ext_vector_type(8)))  float  v8f;

#define D_MODEL 768
#define H_DIM   1536
#define N_TOK   196
#define B_SZ    256
#define N_EXP   3
#define BM      112                 // 7 M-tiles; 50176 = 112 * 448
#define MT_N    (BM / 16)           // 7
#define BK      32
#define KC_N    (D_MODEL / BK)      // 24
#define NT_N    (D_MODEL / 16)      // 48 N-tiles per expert
#define A_CHUNK_DW (MT_N * 32 * 8)  // 1792 dwords per A chunk
#define B_CHUNK_DW (NT_N * 32 * 8)  // 12288 dwords per (expert, kc) chunk
#define B_CHUNK_BYTES (B_CHUNK_DW * 4)  // 49152

__device__ __forceinline__ float gelu_exact(float v) {
  return 0.5f * v * (1.0f + erff(v * 0.70710678118654752f));
}

__device__ __forceinline__ unsigned short f2bf(float f) {
  unsigned u = __builtin_bit_cast(unsigned, f);
  u += 0x7FFFu + ((u >> 16) & 1u);     // round-to-nearest-even
  return (unsigned short)(u >> 16);
}
__device__ __forceinline__ unsigned pack2bf(float a, float b) {
  return (unsigned)f2bf(a) | ((unsigned)f2bf(b) << 16);
}

struct Frag32B { uint4 lo, hi; };
__device__ __forceinline__ v16bf load_frag(const unsigned* p) {
  Frag32B t;
  t.lo = *(const uint4*)(p);
  t.hi = *(const uint4*)(p + 4);
  return __builtin_bit_cast(v16bf, t);
}

// CDNA5 async global->LDS copy, 16B per lane (GVS addressing: saddr base +
// 32-bit vgpr offset; VDST vgpr holds the LDS byte offset).  ASYNCcnt-tracked.
__device__ __forceinline__ void async_copy_b128(unsigned lds_off, unsigned goff,
                                                const void* gbase) {
  asm volatile("global_load_async_to_lds_b128 %0, %1, %2"
               :: "v"(lds_off), "v"(goff), "s"(gbase)
               : "memory");
}
__device__ __forceinline__ void wait_async_le6() {
#if __has_builtin(__builtin_amdgcn_s_wait_asynccnt)
  __builtin_amdgcn_s_wait_asynccnt(6);
#else
  asm volatile("s_wait_asynccnt 0x6" ::: "memory");
#endif
}
__device__ __forceinline__ void wait_async_zero() {
#if __has_builtin(__builtin_amdgcn_s_wait_asynccnt)
  __builtin_amdgcn_s_wait_asynccnt(0);
#else
  asm volatile("s_wait_asynccnt 0x0" ::: "memory");
#endif
}

// ---------------------------------------------------------------------------
// Kernel 1: convert W_ext (E x D x D f32) to bf16, pre-swizzled into the exact
// contiguous-k B-fragment order this file uses for v_wmma_f32_16x16x32_bf16:
//   dword index = (((e*24 + kc)*48 + nt)*32 + lane)*8 + v
//   holds bf16 pair ( W[k][n], W[k+1][n] ) with
//   k = kc*32 + (lane/16)*16 + v*2 ,  n = nt*16 + (lane%16)
// ---------------------------------------------------------------------------
__global__ __launch_bounds__(256) void swizzle_wext(const float* __restrict__ W,
                                                    unsigned* __restrict__ Wb) {
  int idx = blockIdx.x * 256 + threadIdx.x;    // < 3*24*12288 = 884736
  int v    = idx & 7;
  int t    = idx >> 3;
  int lane = t & 31;
  t >>= 5;
  int nt = t % NT_N;  t /= NT_N;
  int kc = t % KC_N;  int e = t / KC_N;
  int k = kc * BK + (lane >> 4) * 16 + v * 2;
  int n = nt * 16 + (lane & 15);
  float w0 = W[((size_t)e * D_MODEL + k) * D_MODEL + n];
  float w1 = W[((size_t)e * D_MODEL + k + 1) * D_MODEL + n];
  Wb[idx] = pack2bf(w0, w1);
}

// ---------------------------------------------------------------------------
// Kernel 2: fused extractor GEMM + bias + LayerNorm + GELU + coverage-weighted
// token reduction.  grid = (50176/112, 3 experts), block = 512 (16 waves).
// Each wave owns ALL 7 M-tiles x 3 N-tiles = 21 f32 fragments (168 VGPRs);
// with 3 resident B fragments + 1 reloaded A fragment the live set stays
// under the 256-VGPR budget -> no spills.  B tiles double-buffer in LDS via
// async global->LDS copies overlapping the WMMA work.
// ---------------------------------------------------------------------------
__global__ __launch_bounds__(512) void extract_gemm(
    const float* __restrict__ x, const unsigned* __restrict__ Wb,
    const float* __restrict__ b_ext, const float* __restrict__ g_ext,
    const float* __restrict__ h_ext, float* __restrict__ sf) {
  __shared__ unsigned sA[A_CHUNK_DW];           // 1792 dw = 7 KB
  __shared__ unsigned sB[2][B_CHUNK_DW];        // 2 x 48 KB
  __shared__ float redS[BM];
  __shared__ float redQ[BM];

  const int tid  = threadIdx.x;
  const int lane = tid & 31;
  const int wave = tid >> 5;       // 0..15; wave owns cols wave*48 .. +47
  const int e    = blockIdx.y;
  const int Rbase = blockIdx.x * BM;

  const unsigned sB_off[2] = { (unsigned)(size_t)(void*)&sB[0][0],
                               (unsigned)(size_t)(void*)&sB[1][0] };
  const char* wsrc = (const char*)(Wb + (size_t)e * KC_N * B_CHUNK_DW);

  if (tid < BM) { redS[tid] = 0.0f; redQ[tid] = 0.0f; }

  v8f acc[MT_N][3];
#pragma unroll
  for (int mt = 0; mt < MT_N; ++mt)
#pragma unroll
    for (int j = 0; j < 3; ++j)
      acc[mt][j] = v8f{0.f, 0.f, 0.f, 0.f, 0.f, 0.f, 0.f, 0.f};

  // ---- prologue: async-fetch B chunk 0 into buffer 0 (6 x 16B per thread) --
#pragma unroll
  for (int i = 0; i < 6; ++i)
    async_copy_b128(sB_off[0] + tid * 16 + i * 8192,
                    (unsigned)(tid * 16 + i * 8192), wsrc);

  for (int kc = 0; kc < KC_N; ++kc) {
    const int cur = kc & 1;
    __syncthreads();   // previous compute done; both LDS buffers consistent

    // ---- async-fetch B chunk kc+1 into the other buffer ----
    if (kc + 1 < KC_N) {
      const char* src = wsrc + (size_t)(kc + 1) * B_CHUNK_BYTES;
#pragma unroll
      for (int i = 0; i < 6; ++i)
        async_copy_b128(sB_off[cur ^ 1] + tid * 16 + i * 8192,
                        (unsigned)(tid * 16 + i * 8192), src);
    }

    // ---- stage A: 112 rows x 32 K, f32 -> bf16, fragment-ordered ----
    const int kbase = kc * BK;
    for (int p = tid; p < A_CHUNK_DW; p += 512) {
      int row = p >> 4;                 // 0..111
      int kp  = p & 15;                 // pair within K-chunk
      float2 xv = *(const float2*)(x + (size_t)(Rbase + row) * D_MODEL + kbase + kp * 2);
      int k2 = kp * 2;
      int v    = (k2 & 15) >> 1;
      int half = k2 >> 4;
      sA[(((row >> 4) * 32) + half * 16 + (row & 15)) * 8 + v] = pack2bf(xv.x, xv.y);
    }
    // prefetch next A chunk (global_prefetch_b8)
    if (kc + 1 < KC_N) {
      int row = tid >> 4;   // rows 0..31 hint; full chunk follows via L2
      __builtin_prefetch(x + (size_t)(Rbase + row) * D_MODEL + kbase + BK, 0, 0);
    }

    // B chunk kc landed? (loads complete in order; <=6 leaves only kc+1's)
    if (kc + 1 < KC_N) wait_async_le6(); else wait_async_zero();
    __syncthreads();

    // ---- WMMA: 21 matrix ops per wave per K-chunk ----
    v16bf bfr[3];
#pragma unroll
    for (int j = 0; j < 3; ++j)
      bfr[j] = load_frag(&sB[cur][((wave * 3 + j) * 32 + lane) * 8]);
#pragma unroll
    for (int mt = 0; mt < MT_N; ++mt) {
      v16bf afrag = load_frag(&sA[(mt * 32 + lane) * 8]);
#pragma unroll
      for (int j = 0; j < 3; ++j)
        acc[mt][j] = __builtin_amdgcn_wmma_f32_16x16x32_bf16(
            false, afrag, false, bfr[j], (short)0, acc[mt][j], false, false);
    }
  }

  // ---- per-lane column parameters ----
  float be[3], ge[3], he[3];
#pragma unroll
  for (int j = 0; j < 3; ++j) {
    int col = (wave * 3 + j) * 16 + (lane & 15);
    be[j] = b_ext[e * D_MODEL + col];
    ge[j] = g_ext[e * D_MODEL + col];
    he[j] = h_ext[e * D_MODEL + col];
  }

  // ---- LayerNorm statistics across the 768 columns of each row ----
#pragma unroll
  for (int mt = 0; mt < MT_N; ++mt) {
#pragma unroll
    for (int r = 0; r < 8; ++r) {
      float s = 0.0f, q = 0.0f;
#pragma unroll
      for (int j = 0; j < 3; ++j) {
        float z = acc[mt][j][r] + be[j];
        s += z; q += z * z;
      }
#pragma unroll
      for (int m = 8; m >= 1; m >>= 1) {   // reduce across 16-lane half
        s += __shfl_xor(s, m);
        q += __shfl_xor(q, m);
      }
      if ((lane & 15) == 0) {
        int rowl = mt * 16 + (lane >> 4) * 8 + r;
        atomicAdd(&redS[rowl], s);
        atomicAdd(&redQ[rowl], q);
      }
    }
  }
  __syncthreads();

  // ---- LN + GELU + coverage-weighted reduction over tokens ----
  const int ssz = (e == 0) ? 4 : ((e == 1) ? 8 : 16);
  const int nw  = N_TOK - ssz + 1;
  const float invw = 1.0f / (float)(nw * ssz);
  const int b0 = Rbase / N_TOK;
  const int bL = (Rbase + BM - 1) / N_TOK;   // at most b0+1

  float w0a[3], w1a[3];
#pragma unroll
  for (int j = 0; j < 3; ++j) { w0a[j] = 0.0f; w1a[j] = 0.0f; }

#pragma unroll
  for (int mt = 0; mt < MT_N; ++mt) {
#pragma unroll
    for (int r = 0; r < 8; ++r) {
      int rowl = mt * 16 + (lane >> 4) * 8 + r;
      int rg = Rbase + rowl;
      int bb = rg / N_TOK;
      int n  = rg % N_TOK;
      int ci = min(min(n + 1, ssz), min(nw, N_TOK - n));
      float cw = (float)ci * invw;
      float mean = redS[rowl] * (1.0f / 768.0f);
      float var  = redQ[rowl] * (1.0f / 768.0f) - mean * mean;
      float rstd = rsqrtf(var + 1e-5f);
#pragma unroll
      for (int j = 0; j < 3; ++j) {
        float z = acc[mt][j][r] + be[j];
        float y = gelu_exact((z - mean) * rstd * ge[j] + he[j]);
        float wy = cw * y;
        if (bb == b0) w0a[j] += wy; else w1a[j] += wy;
      }
    }
  }
#pragma unroll
  for (int j = 0; j < 3; ++j) {    // fold lane halves (same column)
    w0a[j] += __shfl_xor(w0a[j], 16);
    w1a[j] += __shfl_xor(w1a[j], 16);
  }
  if (lane < 16) {
#pragma unroll
    for (int j = 0; j < 3; ++j) {
      int col = (wave * 3 + j) * 16 + lane;
      atomicAdd(&sf[((size_t)b0 * N_EXP + e) * D_MODEL + col], w0a[j]);
      if (bL != b0)
        atomicAdd(&sf[((size_t)bL * N_EXP + e) * D_MODEL + col], w1a[j]);
    }
  }
}

// ---------------------------------------------------------------------------
// Kernel 3: expert FC1: hid = gelu(ln(sf @ W1 + b1))  [grid (256,3), 256 thr]
// ---------------------------------------------------------------------------
__global__ __launch_bounds__(256) void expert_fc1(
    const float* __restrict__ sf, const float* __restrict__ W1,
    const float* __restrict__ b1, const float* __restrict__ g1,
    const float* __restrict__ h1, float* __restrict__ hid) {
  __shared__ float srow[D_MODEL];
  __shared__ float rs[256], rq[256];
  const int b = blockIdx.x, e = blockIdx.y, t = threadIdx.x;
  const float* sr = sf + ((size_t)b * N_EXP + e) * D_MODEL;
  for (int k = t; k < D_MODEL; k += 256) srow[k] = sr[k];
  __syncthreads();
  float v[6];
#pragma unroll
  for (int j = 0; j < 6; ++j) {
    int c = t + j * 256;
    float a = b1[e * H_DIM + c];
    const float* w = W1 + (size_t)e * D_MODEL * H_DIM + c;
    for (int k = 0; k < D_MODEL; ++k) a += srow[k] * w[(size_t)k * H_DIM];
    v[j] = a;
  }
  float s = 0.0f, q = 0.0f;
#pragma unroll
  for (int j = 0; j < 6; ++j) { s += v[j]; q += v[j] * v[j]; }
  rs[t] = s; rq[t] = q; __syncthreads();
  for (int o = 128; o > 0; o >>= 1) {
    if (t < o) { rs[t] += rs[t + o]; rq[t] += rq[t + o]; }
    __syncthreads();
  }
  float mean = rs[0] / (float)H_DIM;
  float var  = rq[0] / (float)H_DIM - mean * mean;
  float rstd = rsqrtf(var + 1e-5f);
  float* hrow = hid + ((size_t)b * N_EXP + e) * H_DIM;
#pragma unroll
  for (int j = 0; j < 6; ++j) {
    int c = t + j * 256;
    float y = (v[j] - mean) * rstd * g1[e * H_DIM + c] + h1[e * H_DIM + c];
    hrow[c] = gelu_exact(y);
  }
}

// ---------------------------------------------------------------------------
// Kernel 4: expert FC2: eo = ln(hid @ W2 + b2)  [grid (256,3), 256 thr]
// ---------------------------------------------------------------------------
__global__ __launch_bounds__(256) void expert_fc2(
    const float* __restrict__ hid, const float* __restrict__ W2,
    const float* __restrict__ b2, const float* __restrict__ g2,
    const float* __restrict__ h2, float* __restrict__ eo) {
  __shared__ float hrow[H_DIM];
  __shared__ float rs[256], rq[256];
  const int b = blockIdx.x, e = blockIdx.y, t = threadIdx.x;
  const float* hr = hid + ((size_t)b * N_EXP + e) * H_DIM;
  for (int k = t; k < H_DIM; k += 256) hrow[k] = hr[k];
  __syncthreads();
  float v[3];
#pragma unroll
  for (int j = 0; j < 3; ++j) {
    int c = t + j * 256;
    float a = b2[e * D_MODEL + c];
    const float* w = W2 + (size_t)e * H_DIM * D_MODEL + c;
    for (int k = 0; k < H_DIM; ++k) a += hrow[k] * w[(size_t)k * D_MODEL];
    v[j] = a;
  }
  float s = 0.0f, q = 0.0f;
#pragma unroll
  for (int j = 0; j < 3; ++j) { s += v[j]; q += v[j] * v[j]; }
  rs[t] = s; rq[t] = q; __syncthreads();
  for (int o = 128; o > 0; o >>= 1) {
    if (t < o) { rs[t] += rs[t + o]; rq[t] += rq[t + o]; }
    __syncthreads();
  }
  float mean = rs[0] / (float)D_MODEL;
  float var  = rq[0] / (float)D_MODEL - mean * mean;
  float rstd = rsqrtf(var + 1e-5f);
  float* er = eo + ((size_t)b * N_EXP + e) * D_MODEL;
#pragma unroll
  for (int j = 0; j < 3; ++j) {
    int c = t + j * 256;
    er[c] = (v[j] - mean) * rstd * g2[e * D_MODEL + c] + h2[e * D_MODEL + c];
  }
}

// ---------------------------------------------------------------------------
// Kernel 5: gate = softmax(gelu(sf0 @ Wg1 + bg1) @ Wg2 + bg2)  [grid 256]
// ---------------------------------------------------------------------------
__global__ __launch_bounds__(256) void gate_kernel(
    const float* __restrict__ sf, const float* __restrict__ Wg1,
    const float* __restrict__ bg1, const float* __restrict__ Wg2,
    const float* __restrict__ bg2, float* __restrict__ gate) {
  __shared__ float srow[D_MODEL];
  __shared__ float t1[D_MODEL / 2];
  __shared__ float lg[N_EXP];
  const int b = blockIdx.x, t = threadIdx.x;
  const float* sr = sf + (size_t)b * N_EXP * D_MODEL;   // scale 0 features
  for (int k = t; k < D_MODEL; k += 256) srow[k] = sr[k];
  __syncthreads();
  for (int c = t; c < D_MODEL / 2; c += 256) {
    float a = bg1[c];
    for (int k = 0; k < D_MODEL; ++k) a += srow[k] * Wg1[(size_t)k * (D_MODEL / 2) + c];
    t1[c] = gelu_exact(a);
  }
  __syncthreads();
  if (t < N_EXP) {
    float a = bg2[t];
    for (int k = 0; k < D_MODEL / 2; ++k) a += t1[k] * Wg2[k * N_EXP + t];
    lg[t] = a;
  }
  __syncthreads();
  if (t == 0) {
    float m = fmaxf(lg[0], fmaxf(lg[1], lg[2]));
    float p0 = expf(lg[0] - m), p1 = expf(lg[1] - m), p2 = expf(lg[2] - m);
    float is = 1.0f / (p0 + p1 + p2);
    gate[b * N_EXP + 0] = p0 * is;
    gate[b * N_EXP + 1] = p1 * is;
    gate[b * N_EXP + 2] = p2 * is;
  }
}

// ---------------------------------------------------------------------------
// Kernel 6: out = gelu(ln(concat(eo_e * gate_e) @ Wf + bf))  [grid 256]
// ---------------------------------------------------------------------------
__global__ __launch_bounds__(256) void fuse_kernel(
    const float* __restrict__ eo, const float* __restrict__ gate,
    const float* __restrict__ Wf, const float* __restrict__ bfv,
    const float* __restrict__ gf, const float* __restrict__ hf,
    float* __restrict__ out) {
  __shared__ float fr[N_EXP * D_MODEL];   // 2304
  __shared__ float rs[256], rq[256];
  const int b = blockIdx.x, t = threadIdx.x;
  for (int i = t; i < N_EXP * D_MODEL; i += 256) {
    int e2 = i / D_MODEL;
    fr[i] = eo[(size_t)b * N_EXP * D_MODEL + i] * gate[b * N_EXP + e2];
  }
  __syncthreads();
  float v[3];
#pragma unroll
  for (int j = 0; j < 3; ++j) {
    int c = t + j * 256;
    float a = bfv[c];
    for (int k = 0; k < N_EXP * D_MODEL; ++k) a += fr[k] * Wf[(size_t)k * D_MODEL + c];
    v[j] = a;
  }
  float s = 0.0f, q = 0.0f;
#pragma unroll
  for (int j = 0; j < 3; ++j) { s += v[j]; q += v[j] * v[j]; }
  rs[t] = s; rq[t] = q; __syncthreads();
  for (int o = 128; o > 0; o >>= 1) {
    if (t < o) { rs[t] += rs[t + o]; rq[t] += rq[t + o]; }
    __syncthreads();
  }
  float mean = rs[0] / (float)D_MODEL;
  float var  = rq[0] / (float)D_MODEL - mean * mean;
  float rstd = rsqrtf(var + 1e-5f);
#pragma unroll
  for (int j = 0; j < 3; ++j) {
    int c = t + j * 256;
    float y = (v[j] - mean) * rstd * gf[c] + hf[c];
    out[(size_t)b * D_MODEL + c] = gelu_exact(y);
  }
}

// ---------------------------------------------------------------------------
extern "C" void kernel_launch(void* const* d_in, const int* in_sizes, int n_in,
                              void* d_out, int out_size, void* d_ws, size_t ws_size,
                              hipStream_t stream) {
  const float* x     = (const float*)d_in[0];
  const float* W_ext = (const float*)d_in[1];
  const float* b_ext = (const float*)d_in[2];
  const float* g_ext = (const float*)d_in[3];
  const float* h_ext = (const float*)d_in[4];
  const float* W1    = (const float*)d_in[5];
  const float* b1    = (const float*)d_in[6];
  const float* g1    = (const float*)d_in[7];
  const float* h1    = (const float*)d_in[8];
  const float* W2    = (const float*)d_in[9];
  const float* b2    = (const float*)d_in[10];
  const float* g2    = (const float*)d_in[11];
  const float* h2    = (const float*)d_in[12];
  const float* Wg1   = (const float*)d_in[13];
  const float* bg1   = (const float*)d_in[14];
  const float* Wg2   = (const float*)d_in[15];
  const float* bg2   = (const float*)d_in[16];
  const float* Wf    = (const float*)d_in[17];
  const float* bfv   = (const float*)d_in[18];
  const float* gf    = (const float*)d_in[19];
  const float* hf    = (const float*)d_in[20];
  float* out = (float*)d_out;

  char* ws = (char*)d_ws;
  size_t off = 0;
  unsigned* Wb = (unsigned*)(ws + off); off += (size_t)N_EXP * KC_N * B_CHUNK_DW * 4; // 3.54 MB
  float* sf   = (float*)(ws + off); off += (size_t)B_SZ * N_EXP * D_MODEL * 4;        // 2.36 MB
  float* hid  = (float*)(ws + off); off += (size_t)B_SZ * N_EXP * H_DIM * 4;          // 4.72 MB
  float* eo   = (float*)(ws + off); off += (size_t)B_SZ * N_EXP * D_MODEL * 4;        // 2.36 MB
  float* gate = (float*)(ws + off); off += (size_t)B_SZ * N_EXP * 4;

  hipMemsetAsync(sf, 0, (size_t)B_SZ * N_EXP * D_MODEL * 4, stream);

  swizzle_wext<<<(N_EXP * KC_N * B_CHUNK_DW) / 256, 256, 0, stream>>>(W_ext, Wb);
  extract_gemm<<<dim3((B_SZ * N_TOK) / BM, N_EXP), 512, 0, stream>>>(
      x, Wb, b_ext, g_ext, h_ext, sf);
  expert_fc1<<<dim3(B_SZ, N_EXP), 256, 0, stream>>>(sf, W1, b1, g1, h1, hid);
  expert_fc2<<<dim3(B_SZ, N_EXP), 256, 0, stream>>>(hid, W2, b2, g2, h2, eo);
  gate_kernel<<<B_SZ, 256, 0, stream>>>(sf, Wg1, bg1, Wg2, bg2, gate);
  fuse_kernel<<<B_SZ, 256, 0, stream>>>(eo, gate, Wf, bfv, gf, hf, out);
}
